// DomDepResidualLayer_9689446220180
// MI455X (gfx1250) — compile-verified, matrix operand order checked by reading, not run on hardware
//
#include <hip/hip_runtime.h>

#define N_TOK 16384
#define EMB   1024
#define DM    128   // MODULE_DIM == MODULE_COUNT

typedef __attribute__((ext_vector_type(16))) __bf16 bf16x16;
typedef __attribute__((ext_vector_type(8)))  float  floatx8;

// float -> bf16 bits, round-to-nearest-even (pure integer ops)
__device__ __forceinline__ unsigned short bfbits(float f) {
    union { float f; unsigned u; } A; A.f = f;
    return (unsigned short)((A.u + 0x7FFFu + ((A.u >> 16) & 1u)) >> 16);
}
__device__ __forceinline__ __bf16 f2bf(float f) {
    union { unsigned short s; __bf16 b; } B; B.s = bfbits(f);
    return B.b;
}

// Load one 16x32 bf16 WMMA A/B fragment from an LDS tile (row stride `ld` bf16, even).
// Wave32 layout (ISA 7.12.2): lanes 0-15 -> K 0..7 & 16..23; lanes 16-31 -> K 8..15 & 24..31.
__device__ __forceinline__ bf16x16 frag_ld(const __bf16* __restrict__ base, int ld, int tile0) {
    const int lane = threadIdx.x & 31;
    const int rc   = tile0 + (lane & 15);
    const int kb   = (lane >> 4) << 3;
    const __bf16* p = base + rc * ld + kb;
    const unsigned* p0 = (const unsigned*)p;
    const unsigned* p1 = (const unsigned*)(p + 16);
    union { bf16x16 v; unsigned u[8]; } U;
    U.u[0] = p0[0]; U.u[1] = p0[1]; U.u[2] = p0[2]; U.u[3] = p0[3];
    U.u[4] = p1[0]; U.u[5] = p1[1]; U.u[6] = p1[2]; U.u[7] = p1[3];
    return U.v;
}

#define WMMA_BF16(a, b, c) \
    __builtin_amdgcn_wmma_f32_16x16x32_bf16(false, (a), false, (b), (short)0, (c), false, false)

#define LDT 40          // LDS tile row stride in bf16: 80B rows, 16B aligned, odd bank stride

// Async copy of one ROWSx32 bf16 chunk (ROWS rows x 64B) global -> LDS.
// B128 wave-instructions tracked on ASYNCcnt. ROWS*4 quads split over 256 threads.
template <int ROWS>
__device__ __forceinline__ void async_chunk(__bf16* dstLds, const __bf16* __restrict__ srcG,
                                            int k0, int tid) {
    constexpr int PER_T = (ROWS * 4) / 256;     // 16B quads per thread
#pragma unroll
    for (int j = 0; j < PER_T; ++j) {
        int q  = tid * PER_T + j;
        int r  = q >> 2;                        // row
        int qb = (q & 3) << 4;                  // byte offset within 64B row chunk
        unsigned ldsa = (unsigned)(uintptr_t)(dstLds + r * LDT) + (unsigned)qb;
        const char* g = (const char*)(srcG + (long)r * DM + k0) + qb;
        asm volatile("global_load_async_to_lds_b128 %0, %1, off"
                     :: "v"(ldsa), "v"(g) : "memory");
    }
}

// K=128 GEMM mainloop, both operands bf16 in global (row stride 128 = DM).
// RT = 16-row tiles per wave (A rows per block = 128*RT). ASYNCcnt double-buffered.
// Per chunk per wave: RT A-frags + 8 B-frags feed RT*8 WMMAs (B reused across RT).
template <int RT>
__device__ __forceinline__ void gemm128_async(const __bf16* __restrict__ Ag,
                                              const __bf16* __restrict__ Bg,
                                              __bf16* Asm, __bf16* Bsm,
                                              floatx8 (&acc)[RT][8], int tid, int wave) {
    constexpr int AROWS = 128 * RT;
    constexpr int AT = AROWS * LDT;             // A tile elems per buffer
    constexpr int BT = 128 * LDT;
    async_chunk<AROWS>(Asm, Ag, 0, tid);
    async_chunk<128>  (Bsm, Bg, 0, tid);
    for (int c = 0; c < 4; ++c) {
        __bf16* Ac = Asm + (c & 1) * AT;
        __bf16* Bc = Bsm + (c & 1) * BT;
        if (c < 3) {
            async_chunk<AROWS>(Asm + ((c + 1) & 1) * AT, Ag, (c + 1) * 32, tid);
            async_chunk<128>  (Bsm + ((c + 1) & 1) * BT, Bg, (c + 1) * 32, tid);
            if (RT == 1) asm volatile("s_wait_asynccnt 0x4" ::: "memory");
            else         asm volatile("s_wait_asynccnt 0x6" ::: "memory");
        } else {
            asm volatile("s_wait_asynccnt 0x0" ::: "memory");
        }
        __syncthreads();
        bf16x16 a[RT];
#pragma unroll
        for (int rt = 0; rt < RT; ++rt) a[rt] = frag_ld(Ac, LDT, (wave * RT + rt) * 16);
#pragma unroll
        for (int ct = 0; ct < 8; ++ct) {
            bf16x16 b = frag_ld(Bc, LDT, ct * 16);
#pragma unroll
            for (int rt = 0; rt < RT; ++rt)
                acc[rt][ct] = WMMA_BF16(a[rt], b, acc[rt][ct]);
        }
        __syncthreads();   // protects double-buffer reuse (WAR)
    }
}

// ---------------------------------------------------------------------------
// Kernel 0: f32 -> bf16 bulk convert (pairs -> packed u32 stores)
// ---------------------------------------------------------------------------
__global__ void __launch_bounds__(256) k_cvt(const float* __restrict__ src,
                                             __bf16* __restrict__ dst, int n_pairs) {
    int i = blockIdx.x * 256 + threadIdx.x;
    if (i < n_pairs) {
        float2 f = ((const float2*)src)[i];
        unsigned u = (unsigned)bfbits(f.x) | ((unsigned)bfbits(f.y) << 16);
        ((unsigned*)dst)[i] = u;
    }
}

// ---------------------------------------------------------------------------
// Kernel 1: enc = h0 @ we0_w^T + we0_b  -> bf16 [N, 128]
// h0 is f32 (must convert in-kernel); K=1024 in 32-wide chunks. Memory-bound.
// ---------------------------------------------------------------------------
__global__ void __launch_bounds__(256) k_enc(const float* __restrict__ h0,
                                             const float* __restrict__ we0_w,
                                             const float* __restrict__ we0_b,
                                             __bf16* __restrict__ enc_bf) {
    __shared__ __align__(16) __bf16 As[128 * 34];
    __shared__ __align__(16) __bf16 Bs[128 * 34];
    __shared__ __align__(16) __bf16 Eout[128 * 128];
    const int tid  = threadIdx.x;
    const int wave = tid >> 5;
    const int lane = tid & 31;
    const long tok0 = (long)blockIdx.x * 128;

    floatx8 acc[8];
#pragma unroll
    for (int t = 0; t < 8; ++t) acc[t] = (floatx8)0.0f;

    for (int e0 = 0; e0 < EMB; e0 += 32) {
        for (int i = tid; i < 128 * 32; i += 256) {
            int r = i >> 5, k = i & 31;
            As[r * 34 + k] = f2bf(h0[(tok0 + r) * EMB + e0 + k]);
        }
        for (int i = tid; i < 128 * 32; i += 256) {
            int c = i >> 5, k = i & 31;
            Bs[c * 34 + k] = f2bf(we0_w[(long)c * EMB + e0 + k]);
        }
        __syncthreads();
        bf16x16 a = frag_ld(As, 34, wave * 16);
#pragma unroll
        for (int ct = 0; ct < 8; ++ct) {
            bf16x16 b = frag_ld(Bs, 34, ct * 16);
            acc[ct] = WMMA_BF16(a, b, acc[ct]);
        }
        __syncthreads();
    }
    const int n = lane & 15, mb = (lane >> 4) << 3;
#pragma unroll
    for (int ct = 0; ct < 8; ++ct) {
        int col = ct * 16 + n;
        float bias = we0_b[col];
#pragma unroll
        for (int r = 0; r < 8; ++r) {
            int row = wave * 16 + mb + r;
            Eout[row * 128 + col] = f2bf(acc[ct][r] + bias);
        }
    }
    __syncthreads();
    unsigned* gout = (unsigned*)enc_bf + tok0 * 64;
    const unsigned* sout = (const unsigned*)Eout;
    for (int i = tid; i < 128 * 64; i += 256) gout[i] = sout[i];
}

// ---------------------------------------------------------------------------
// Kernel 2: gl = enc @ ws_w^T + ws_b  (f32 [N, 128]) ; fully async-staged
// ---------------------------------------------------------------------------
__global__ void __launch_bounds__(256) k_gl(const __bf16* __restrict__ enc_bf,
                                            const __bf16* __restrict__ ws_bf,
                                            const float* __restrict__ ws_b,
                                            float* __restrict__ gl) {
    __shared__ __align__(16) __bf16 Asm[2 * 128 * LDT];
    __shared__ __align__(16) __bf16 Bsm[2 * 128 * LDT];
    const int tid  = threadIdx.x;
    const int wave = tid >> 5;
    const int lane = tid & 31;
    const long tok0 = (long)blockIdx.x * 128;

    floatx8 acc[1][8];
#pragma unroll
    for (int t = 0; t < 8; ++t) acc[0][t] = (floatx8)0.0f;

    gemm128_async<1>(enc_bf + tok0 * DM, ws_bf, Asm, Bsm, acc, tid, wave);

    const int n = lane & 15, mb = (lane >> 4) << 3;
#pragma unroll
    for (int ct = 0; ct < 8; ++ct) {
        int col = ct * 16 + n;
        float bias = ws_b[col];
#pragma unroll
        for (int r = 0; r < 8; ++r) {
            long tokg = tok0 + wave * 16 + mb + r;
            gl[tokg * DM + col] = acc[0][ct][r] + bias;
        }
    }
}

// ---------------------------------------------------------------------------
// Kernel 3: in-place softmax over axis 0 (16384 tokens), per column m.
// ---------------------------------------------------------------------------
__global__ void __launch_bounds__(256) k_softmax(float* __restrict__ gl) {
    const int m = blockIdx.x;
    const int tid = threadIdx.x;
    __shared__ float red[256];

    float mx = -3.402823466e38f;
    for (int n = tid; n < N_TOK; n += 256) mx = fmaxf(mx, gl[(long)n * DM + m]);
    red[tid] = mx; __syncthreads();
    for (int s = 128; s > 0; s >>= 1) {
        if (tid < s) red[tid] = fmaxf(red[tid], red[tid + s]);
        __syncthreads();
    }
    mx = red[0]; __syncthreads();

    float sum = 0.0f;
    for (int n = tid; n < N_TOK; n += 256) sum += __expf(gl[(long)n * DM + m] - mx);
    red[tid] = sum; __syncthreads();
    for (int s = 128; s > 0; s >>= 1) {
        if (tid < s) red[tid] += red[tid + s];
        __syncthreads();
    }
    float inv = 1.0f / red[0]; __syncthreads();

    for (int n = tid; n < N_TOK; n += 256) {
        long idx = (long)n * DM + m;
        gl[idx] = __expf(gl[idx] - mx) * inv;
    }
}

// ---------------------------------------------------------------------------
// Kernel 4 (hot, 68.7 GFLOP): per (256-token tile, expert m):
//   acts = relu(enc_tile @ wm_w[m]^T + wm_b[m])   -- accumulators only
//   mixed[n, m] = sum_k acts[n,k] * gate[n,k]     -- shfl_xor butterfly
// 2x8 accumulator tiles per wave: 16 WMMAs per 2 A-frags + 8 B-frags (5 ds/WMMA).
// Both operand tiles arrive via global_load_async_to_lds_b128 (ASYNCcnt).
// ---------------------------------------------------------------------------
__global__ void __launch_bounds__(256) k_experts(const __bf16* __restrict__ enc_bf,
                                                 const __bf16* __restrict__ wm_bf,
                                                 const float* __restrict__ wm_b,
                                                 const float* __restrict__ gate,
                                                 __bf16* __restrict__ mixed_bf) {
    __shared__ __align__(16) __bf16 Asm[2 * 256 * LDT];   // 40 KB
    __shared__ __align__(16) __bf16 Bsm[2 * 128 * LDT];   // 20 KB
    const int tid  = threadIdx.x;
    const int wave = tid >> 5;
    const int lane = tid & 31;
    const long tok0 = (long)blockIdx.x * 256;
    const int  m    = blockIdx.y;

    floatx8 acc[2][8];
#pragma unroll
    for (int rt = 0; rt < 2; ++rt)
#pragma unroll
        for (int t = 0; t < 8; ++t) acc[rt][t] = (floatx8)0.0f;

    gemm128_async<2>(enc_bf + tok0 * DM, wm_bf + (long)m * DM * DM, Asm, Bsm, acc, tid, wave);

    // fused bias + ReLU + gate-weighted reduction over the 128 acts columns
    const int n = lane & 15, mb = (lane >> 4) << 3;
#pragma unroll
    for (int rt = 0; rt < 2; ++rt) {
        float part[8];
#pragma unroll
        for (int r = 0; r < 8; ++r) part[r] = 0.0f;
#pragma unroll
        for (int ct = 0; ct < 8; ++ct) {
            int col = ct * 16 + n;
            float bias = wm_b[m * DM + col];
#pragma unroll
            for (int r = 0; r < 8; ++r) {
                long tokg = tok0 + (wave * 2 + rt) * 16 + mb + r;
                float a_ = fmaxf(acc[rt][ct][r] + bias, 0.0f);
                part[r] += a_ * gate[tokg * DM + col];
            }
        }
#pragma unroll
        for (int r = 0; r < 8; ++r) {
            float v = part[r];
            v += __shfl_xor(v, 1, 32);
            v += __shfl_xor(v, 2, 32);
            v += __shfl_xor(v, 4, 32);
            v += __shfl_xor(v, 8, 32);   // sum within each 16-lane half
            if ((lane & 15) == 0) {
                long tokg = tok0 + (wave * 2 + rt) * 16 + mb + r;
                mixed_bf[tokg * DM + m] = f2bf(v);
            }
        }
    }
}

// ---------------------------------------------------------------------------
// Kernel 5: out = relu(mixed @ we1_w^T + we1_b + h0)   [N, 1024]
// 256-token x 128-embed-col blocks, 2x8 accumulator tiles per wave.
// ---------------------------------------------------------------------------
__global__ void __launch_bounds__(256) k_decode(const __bf16* __restrict__ mixed_bf,
                                                const __bf16* __restrict__ we1_bf,
                                                const float* __restrict__ we1_b,
                                                const float* __restrict__ h0,
                                                float* __restrict__ out) {
    __shared__ __align__(16) __bf16 Asm[2 * 256 * LDT];
    __shared__ __align__(16) __bf16 Bsm[2 * 128 * LDT];
    const int tid  = threadIdx.x;
    const int wave = tid >> 5;
    const int lane = tid & 31;
    const long tok0 = (long)blockIdx.x * 256;
    const int  col0 = blockIdx.y * 128;

    floatx8 acc[2][8];
#pragma unroll
    for (int rt = 0; rt < 2; ++rt)
#pragma unroll
        for (int t = 0; t < 8; ++t) acc[rt][t] = (floatx8)0.0f;

    gemm128_async<2>(mixed_bf + tok0 * DM, we1_bf + (long)col0 * DM, Asm, Bsm, acc, tid, wave);

    const int n = lane & 15, mb = (lane >> 4) << 3;
#pragma unroll
    for (int rt = 0; rt < 2; ++rt) {
#pragma unroll
        for (int ct = 0; ct < 8; ++ct) {
            int colg = col0 + ct * 16 + n;
            float bias = we1_b[colg];
#pragma unroll
            for (int r = 0; r < 8; ++r) {
                long tokg = tok0 + (wave * 2 + rt) * 16 + mb + r;
                float v = acc[rt][ct][r] + bias + h0[tokg * EMB + colg];
                out[tokg * EMB + colg] = fmaxf(v, 0.0f);
            }
        }
    }
}

// ---------------------------------------------------------------------------
extern "C" void kernel_launch(void* const* d_in, const int* in_sizes, int n_in,
                              void* d_out, int out_size, void* d_ws, size_t ws_size,
                              hipStream_t stream) {
    (void)in_sizes; (void)n_in; (void)out_size; (void)ws_size;
    const float* h0    = (const float*)d_in[0];
    const float* we0_w = (const float*)d_in[1];
    const float* we0_b = (const float*)d_in[2];
    const float* ws_w  = (const float*)d_in[3];
    const float* ws_b  = (const float*)d_in[4];
    const float* wm_w  = (const float*)d_in[5];
    const float* wm_b  = (const float*)d_in[6];
    const float* we1_w = (const float*)d_in[7];
    const float* we1_b = (const float*)d_in[8];
    float* out = (float*)d_out;

    // workspace: enc_bf 4M | gl f32 8M | mixed_bf 4M | wm_bf 4M | we1_bf 256K | ws_bf 32K
    char* ws = (char*)d_ws;
    size_t off = 0;
    __bf16* enc_bf   = (__bf16*)(ws + off); off += (size_t)N_TOK * DM * 2;
    float*  gl       = (float*) (ws + off); off += (size_t)N_TOK * DM * 4;
    __bf16* mixed_bf = (__bf16*)(ws + off); off += (size_t)N_TOK * DM * 2;
    __bf16* wm_bf    = (__bf16*)(ws + off); off += (size_t)DM * DM * DM * 2;
    __bf16* we1_bf   = (__bf16*)(ws + off); off += (size_t)EMB * DM * 2;
    __bf16* ws_bf    = (__bf16*)(ws + off); off += (size_t)DM * DM * 2;

    dim3 blk(256);
    const int p_wm  = DM * DM * DM / 2, p_we1 = EMB * DM / 2, p_ws = DM * DM / 2;
    k_cvt<<<dim3((p_wm  + 255) / 256), blk, 0, stream>>>(wm_w,  wm_bf,  p_wm);
    k_cvt<<<dim3((p_we1 + 255) / 256), blk, 0, stream>>>(we1_w, we1_bf, p_we1);
    k_cvt<<<dim3((p_ws  + 255) / 256), blk, 0, stream>>>(ws_w,  ws_bf,  p_ws);

    k_enc    <<<dim3(N_TOK / 128),            blk, 0, stream>>>(h0, we0_w, we0_b, enc_bf);
    k_gl     <<<dim3(N_TOK / 128),            blk, 0, stream>>>(enc_bf, ws_bf, ws_b, gl);
    k_softmax<<<dim3(DM),                     blk, 0, stream>>>(gl);
    k_experts<<<dim3(N_TOK / 256, DM),        blk, 0, stream>>>(enc_bf, wm_bf, wm_b, gl, mixed_bf);
    k_decode <<<dim3(N_TOK / 256, EMB / 128), blk, 0, stream>>>(mixed_bf, we1_bf, we1_b, h0, out);
}
